// PointNetSeg_7705171329405
// MI455X (gfx1250) — matrix operand order repaired and checked by reading, not compiled, not run on hardware
//
#include <hip/hip_runtime.h>
#include <hip/hip_bf16.h>

// ---------------------------------------------------------------------------
// PointNet++ part-seg forward for gfx1250 (MI455X).
// GEMM layers run on v_wmma_f32_16x16x32_bf16 (wave32, one 16x16 tile/wave),
// fp32 accumulate, BN batch-stats via global fp32 atomics.
// Required workspace: ~75 MB (checked against ws_size before launching).
// ---------------------------------------------------------------------------

#define B_    4
#define N_    16384
#define FEAT_ 6
#define NCLS_ 13

typedef __attribute__((ext_vector_type(16))) __bf16 bf16x16;
typedef __attribute__((ext_vector_type(8)))  __bf16 bf16x8;
typedef __attribute__((ext_vector_type(8)))  float  f32x8;

static inline int cdiv(long long a, long long b) { return (int)((a + b - 1) / b); }

// -------------------------------- utility ----------------------------------
__global__ void k_zero(float* __restrict__ p, int n) {
  int t = blockIdx.x * blockDim.x + threadIdx.x;
  if (t < n) p[t] = 0.0f;
}

// fp32 weight (Cout x Cin, row major) -> bf16 padded (Cout x Kpad), zeros in pad
__global__ void k_convert_w(const float* __restrict__ w, __bf16* __restrict__ out,
                            int Cout, int Cin, int Kpad) {
  int t = blockIdx.x * blockDim.x + threadIdx.x;
  if (t >= Cout * Kpad) return;
  int co = t / Kpad, kk = t % Kpad;
  out[t] = (kk < Cin) ? (__bf16)w[co * Cin + kk] : (__bf16)0.0f;
}

// ------------------------------- FPS ---------------------------------------
// One block per batch. dists persists in global ws. Matches reference:
// idx[0]=0, argmax ties -> lowest index.
__global__ void k_fps(const float* __restrict__ p, float* __restrict__ dists,
                      int* __restrict__ idx, int n, int npoint) {
  int b = blockIdx.x, tid = threadIdx.x;
  const float* pb = p + (size_t)b * n * 3;
  float* db = dists + (size_t)b * n;
  int* ib = idx + (size_t)b * npoint;
  __shared__ float sval[256];
  __shared__ int   sidx[256];
  __shared__ float lastP[3];
  for (int j = tid; j < n; j += 256) db[j] = 1e10f;
  if (tid == 0) { ib[0] = 0; lastP[0] = pb[0]; lastP[1] = pb[1]; lastP[2] = pb[2]; }
  __syncthreads();
  for (int i = 1; i < npoint; ++i) {
    float lx = lastP[0], ly = lastP[1], lz = lastP[2];
    float bv = -1.0f; int bj = 0x7fffffff;
    for (int j = tid; j < n; j += 256) {
      float dx = pb[j*3] - lx, dy = pb[j*3+1] - ly, dz = pb[j*3+2] - lz;
      float d = dx*dx + dy*dy + dz*dz;
      float dm = db[j]; dm = dm < d ? dm : d; db[j] = dm;
      if (dm > bv || (dm == bv && j < bj)) { bv = dm; bj = j; }
    }
    sval[tid] = bv; sidx[tid] = bj; __syncthreads();
    for (int s = 128; s > 0; s >>= 1) {
      if (tid < s) {
        float ov = sval[tid+s]; int oj = sidx[tid+s];
        if (ov > sval[tid] || (ov == sval[tid] && oj < sidx[tid])) { sval[tid] = ov; sidx[tid] = oj; }
      }
      __syncthreads();
    }
    if (tid == 0) {
      int nx = sidx[0]; ib[i] = nx;
      lastP[0] = pb[nx*3]; lastP[1] = pb[nx*3+1]; lastP[2] = pb[nx*3+2];
    }
    __syncthreads();
  }
}

__global__ void k_gather_centers(const float* __restrict__ p, const int* __restrict__ idx,
                                 float* __restrict__ pc, int n, int npoint) {
  int t = blockIdx.x * blockDim.x + threadIdx.x;
  if (t >= B_ * npoint) return;
  int b = t / npoint;
  int j = idx[t];
  const float* src = p + ((size_t)b * n + j) * 3;
  pc[t*3+0] = src[0]; pc[t*3+1] = src[1]; pc[t*3+2] = src[2];
}

// ------------------------------ kNN + group --------------------------------
// One block per (batch, center). 32 rounds of lexicographic argmin on (d, j)
// with a moving threshold (= top_k ascending with index tie-break), then the
// block writes the grouped bf16 rows [rel_xyz | feat | zero pad] for WMMA.
__global__ void k_knn_group(const float* __restrict__ p, const float* __restrict__ x,
                            const float* __restrict__ pc, __bf16* __restrict__ act,
                            int n, int npoint, int nsample, int Cin, int Kpad) {
  int bc = blockIdx.x, tid = threadIdx.x;
  int b = bc / npoint, c = bc % npoint;
  const float* pb = p + (size_t)b * n * 3;
  const float* ce = pc + ((size_t)b * npoint + c) * 3;
  float cx = ce[0], cy = ce[1], cz = ce[2];
  __shared__ float sval[256];
  __shared__ int   sidx[256];
  __shared__ int   sel[64];
  __shared__ float shLastD;
  __shared__ int   shLastJ;
  if (tid == 0) { shLastD = -1.0f; shLastJ = -1; }
  __syncthreads();
  for (int s = 0; s < nsample; ++s) {
    float lastD = shLastD; int lastJ = shLastJ;
    float bv = 3.4e38f; int bj = 0x7fffffff;
    for (int j = tid; j < n; j += 256) {
      float dx = pb[j*3] - cx, dy = pb[j*3+1] - cy, dz = pb[j*3+2] - cz;
      float d = dx*dx + dy*dy + dz*dz;
      bool after = (d > lastD) || (d == lastD && j > lastJ);
      if (after && (d < bv || (d == bv && j < bj))) { bv = d; bj = j; }
    }
    sval[tid] = bv; sidx[tid] = bj; __syncthreads();
    for (int st = 128; st > 0; st >>= 1) {
      if (tid < st) {
        float ov = sval[tid+st]; int oj = sidx[tid+st];
        if (ov < sval[tid] || (ov == sval[tid] && oj < sidx[tid])) { sval[tid] = ov; sidx[tid] = oj; }
      }
      __syncthreads();
    }
    if (tid == 0) { sel[s] = sidx[0]; shLastD = sval[0]; shLastJ = sidx[0]; }
    __syncthreads();
  }
  size_t rowBase = ((size_t)b * npoint + c) * (size_t)nsample;
  int total = nsample * Kpad;
  for (int t = tid; t < total; t += 256) {
    int s = t / Kpad, kk = t % Kpad;
    int j = sel[s];
    float v;
    if (kk < 3)            v = pb[j*3 + kk] - ce[kk];
    else if (kk < 3 + Cin) v = x[((size_t)b * n + j) * Cin + (kk - 3)];
    else                   v = 0.0f;
    act[(rowBase + s) * Kpad + kk] = (__bf16)v;
  }
}

// ------------------------------- WMMA GEMM ---------------------------------
// C(MxN) = A(MxKpad, bf16) * W^T (W stored N x Kpad bf16, row major).
// One 16x16 tile per wave, 4 waves/block. Accumulates per-channel sum/sumsq
// for BatchNorm batch statistics via fp32 global atomics.
// A frag (16-bit A 16x32): lane l -> row (l&15),
//   K = (l>>4)*16 .. +15 contiguous (matches B).
// B frag (same k order as A, not AMD's table order): lane l -> col (l&15),
//   K = (l>>4)*16 .. +15 contiguous -> one 32B load of W[n][k..k+15].
// C: VGPR r / lane l -> (M = r + 8*(l>>4), N = l&15).
__global__ void __launch_bounds__(128)
k_gemm_bf16(const __bf16* __restrict__ A, const __bf16* __restrict__ Wb,
            float* __restrict__ C, float* __restrict__ stats,
            int M, int N, int Kpad) {
  int lane = threadIdx.x & 31;
  int wave = threadIdx.x >> 5;
  int tilesN = N >> 4;
  int tile = blockIdx.x * 4 + wave;
  if (tile >= (M >> 4) * tilesN) return;       // wave-uniform -> scalar branch
  int m0 = (tile / tilesN) << 4;
  int n0 = (tile % tilesN) << 4;
  int half = lane >> 4;
  int sub  = lane & 15;
  const __bf16* arow = A  + (size_t)(m0 + sub) * Kpad + half * 16;
  const __bf16* brow = Wb + (size_t)(n0 + sub) * Kpad + half * 16;
  f32x8 acc = {};
  for (int k = 0; k < Kpad; k += 32) {
    __builtin_prefetch(arow + k + 128, 0, 0);  // global_prefetch_b8 (speculative)
    bf16x8  alo = *reinterpret_cast<const bf16x8*>(arow + k);
    bf16x8  ahi = *reinterpret_cast<const bf16x8*>(arow + k + 8);
    bf16x16 bV  = *reinterpret_cast<const bf16x16*>(brow + k);
    bf16x16 aV;
#pragma unroll
    for (int i = 0; i < 8; ++i) { aV[i] = alo[i]; aV[i + 8] = ahi[i]; }
    acc = __builtin_amdgcn_wmma_f32_16x16x32_bf16(false, aV, false, bV,
                                                  (short)0, acc, false, false);
  }
  float s = 0.0f, s2 = 0.0f;
#pragma unroll
  for (int r = 0; r < 8; ++r) {
    float v = acc[r];
    C[(size_t)(m0 + r + half * 8) * N + (n0 + sub)] = v;
    s += v; s2 += v * v;
  }
  atomicAdd(&stats[n0 + sub], s);
  atomicAdd(&stats[N + n0 + sub], s2);
}

// ------------------------------ BatchNorm ----------------------------------
__global__ void k_bn_finalize(float* __restrict__ st, int N, int M) {
  int c = blockIdx.x * blockDim.x + threadIdx.x;
  if (c >= N) return;
  float mean = st[c] / (float)M;
  float var  = st[N + c] / (float)M - mean * mean;
  var = var > 0.0f ? var : 0.0f;
  st[c] = mean;
  st[N + c] = rsqrtf(var + 1e-5f);
}

__global__ void k_bn_to_bf16(const float* __restrict__ X, const float* __restrict__ st,
                             __bf16* __restrict__ out, int M, int N, int Kpad, int relu) {
  long long t = (long long)blockIdx.x * blockDim.x + threadIdx.x;
  if (t >= (long long)M * Kpad) return;
  int kk = (int)(t % Kpad);
  long long r = t / Kpad;
  float v = 0.0f;
  if (kk < N) {
    float x = X[r * N + kk];
    v = (x - st[kk]) * st[N + kk];
    if (relu) v = v > 0.0f ? v : 0.0f;
  }
  out[t] = (__bf16)v;
}

__global__ void k_bn_to_f32(const float* __restrict__ X, const float* __restrict__ st,
                            float* __restrict__ out, int M, int N) {
  long long t = (long long)blockIdx.x * blockDim.x + threadIdx.x;
  if (t >= (long long)M * N) return;
  int c = (int)(t % N);
  float v = (X[t] - st[c]) * st[N + c];
  out[t] = v > 0.0f ? v : 0.0f;   // bn_relu
}

// ------------------------------ pooling ------------------------------------
__global__ void k_maxpool(const float* __restrict__ X, float* __restrict__ out,
                          int BP, int ns, int C) {
  int t = blockIdx.x * blockDim.x + threadIdx.x;
  if (t >= BP * C) return;
  int bp = t / C, c = t % C;
  const float* base = X + (size_t)bp * ns * C + c;
  float m = -3.4e38f;
  for (int s = 0; s < ns; ++s) { float v = base[(size_t)s * C]; m = v > m ? v : m; }
  out[t] = m;
}

// ------------------------- 3-NN interpolation + concat ---------------------
__global__ void k_interp_concat(const float* __restrict__ p1, const float* __restrict__ x1,
                                const float* __restrict__ p2, const float* __restrict__ x2,
                                __bf16* __restrict__ act,
                                int n1, int n2, int C1, int C2, int Kpad) {
  int t = blockIdx.x * blockDim.x + threadIdx.x;
  if (t >= B_ * n1) return;
  int b = t / n1;
  const float* q = p1 + (size_t)t * 3;
  float qx = q[0], qy = q[1], qz = q[2];
  const float* pb = p2 + (size_t)b * n2 * 3;
  float d0 = 3.4e38f, d1 = 3.4e38f, d2v = 3.4e38f;
  int   i0 = 0,       i1 = 0,       i2 = 0;
  for (int j = 0; j < n2; ++j) {
    float dx = pb[j*3] - qx, dy = pb[j*3+1] - qy, dz = pb[j*3+2] - qz;
    float d = dx*dx + dy*dy + dz*dz;
    if (d < d0)       { d2v = d1; i2 = i1; d1 = d0; i1 = i0; d0 = d; i0 = j; }
    else if (d < d1)  { d2v = d1; i2 = i1; d1 = d;  i1 = j; }
    else if (d < d2v) { d2v = d;  i2 = j; }
  }
  float w0 = 1.0f / (d0 + 1e-8f), w1 = 1.0f / (d1 + 1e-8f), w2 = 1.0f / (d2v + 1e-8f);
  float ws = w0 + w1 + w2; w0 /= ws; w1 /= ws; w2 /= ws;
  __bf16* row = act + (size_t)t * Kpad;
  if (C1 > 0) {
    const float* x1r = x1 + (size_t)t * C1;
    for (int c = 0; c < C1; ++c) row[c] = (__bf16)x1r[c];
  }
  const float* xb = x2 + (size_t)b * n2 * C2;
  for (int c = 0; c < C2; ++c) {
    float v = w0 * xb[(size_t)i0 * C2 + c] + w1 * xb[(size_t)i1 * C2 + c]
            + w2 * xb[(size_t)i2 * C2 + c];
    row[C1 + c] = (__bf16)v;
  }
  for (int c = C1 + C2; c < Kpad; ++c) row[c] = (__bf16)0.0f;
}

// ------------------------------- seg head ----------------------------------
__global__ void k_head_out(const __bf16* __restrict__ act, const float* __restrict__ w2,
                           const float* __restrict__ b2, float* __restrict__ out, int rows) {
  int t = blockIdx.x * blockDim.x + threadIdx.x;
  if (t >= rows * NCLS_) return;
  int r = t / NCLS_, c = t % NCLS_;
  const __bf16* a = act + (size_t)r * 128;
  const float*  w = w2 + (size_t)c * 128;
  float s = b2[c];
  for (int k = 0; k < 128; ++k) s += (float)a[k] * w[k];
  out[t] = s;
}

// ===========================================================================
extern "C" void kernel_launch(void* const* d_in, const int* in_sizes, int n_in,
                              void* d_out, int out_size, void* d_ws, size_t ws_size,
                              hipStream_t stream) {
  (void)in_sizes; (void)n_in; (void)out_size;
  const float* coord = (const float*)d_in[0];
  const float* feat  = (const float*)d_in[1];
  const float* W[22];
  { int wi = 0; for (int i = 2; i <= 22; ++i) W[wi++] = (const float*)d_in[i];
    W[21] = (const float*)d_in[23]; }                      // head_w1
  const float* head_w2 = (const float*)d_in[25];
  const float* head_b2 = (const float*)d_in[26];
  // head_b1 (d_in[24]) cancels exactly under batch-stat BN -> skipped.

  // layer table: (Cout, Cin, Kpad); Kpad multiple of 32, Cout multiple of 16
  static const int LCO[22] = {32,32,64, 64,64,128, 128,128,256, 256,256,512,
                              256,256, 256,256, 256,128, 128,128,128, 128};
  static const int LCI[22] = {9,32,32, 67,64,64, 131,128,128, 259,256,256,
                              768,256, 384,256, 320,256, 128,128,128, 128};
  static const int LKP[22] = {32,32,32, 96,64,64, 160,128,128, 288,256,256,
                              768,256, 384,256, 320,256, 128,128,128, 128};

  // ---------------- workspace arena ----------------
  char* base = (char*)d_ws; size_t off = 0;
  auto take = [&](size_t bytes) -> char* {
    off = (off + 255) & ~(size_t)255;
    char* p = base + off; off += bytes; return p;
  };
  float* stats = (float*)take(22 * 1024 * sizeof(float));
  float* fpsd  = (float*)take((size_t)B_ * N_ * sizeof(float));
  int* idxL[4]; float* pL[4]; float* xL[4];
  const int NP[4] = {1024, 256, 64, 16};
  const int XC[4] = {64, 128, 256, 512};
  for (int s = 0; s < 4; ++s) {
    idxL[s] = (int*)  take((size_t)B_ * NP[s] * sizeof(int));
    pL[s]   = (float*)take((size_t)B_ * NP[s] * 3 * sizeof(float));
    xL[s]   = (float*)take((size_t)B_ * NP[s] * XC[s] * sizeof(float));
  }
  float* x3n = (float*)take((size_t)B_ * 64   * 256 * sizeof(float));
  float* x2n = (float*)take((size_t)B_ * 256  * 256 * sizeof(float));
  float* x1n = (float*)take((size_t)B_ * 1024 * 128 * sizeof(float));
  __bf16* Wb[22];
  for (int l = 0; l < 22; ++l) Wb[l] = (__bf16*)take((size_t)LCO[l] * LKP[l] * sizeof(__bf16));
  const size_t ACT_EL = 8388608;                            // 65536*128 / 131072*64
  __bf16* actA  = (__bf16*)take(ACT_EL * sizeof(__bf16));
  __bf16* actB  = (__bf16*)take(ACT_EL * sizeof(__bf16));
  float*  outF  = (float*) take(ACT_EL * sizeof(float));
  if (off > ws_size) return;                                // insufficient scratch

  // ---------------- launch helpers ----------------
  auto elt = [&](int n) { return dim3(cdiv(n, 256)); };
  auto run_gemm = [&](const __bf16* in, int l, int M) {
    int Cout = LCO[l], Kp = LKP[l];
    float* st = stats + l * 1024;
    int tiles = (M / 16) * (Cout / 16);
    k_gemm_bf16<<<dim3(cdiv(tiles, 4)), dim3(128), 0, stream>>>(in, Wb[l], outF, st, M, Cout, Kp);
    k_bn_finalize<<<elt(Cout), dim3(256), 0, stream>>>(st, Cout, M);
  };
  auto bn_bf16 = [&](int l, __bf16* dst, int M, int KpadNext, int relu) {
    long long tot = (long long)M * KpadNext;
    k_bn_to_bf16<<<dim3(cdiv(tot, 256)), dim3(256), 0, stream>>>(outF, stats + l * 1024,
                                                                 dst, M, LCO[l], KpadNext, relu);
  };
  auto bn_f32 = [&](int l, float* dst, int M) {
    long long tot = (long long)M * LCO[l];
    k_bn_to_f32<<<dim3(cdiv(tot, 256)), dim3(256), 0, stream>>>(outF, stats + l * 1024, dst, M, LCO[l]);
  };

  // ---------------- pipeline ----------------
  k_zero<<<elt(22 * 1024), dim3(256), 0, stream>>>(stats, 22 * 1024);
  for (int l = 0; l < 22; ++l)
    k_convert_w<<<elt(LCO[l] * LKP[l]), dim3(256), 0, stream>>>(W[l], Wb[l], LCO[l], LCI[l], LKP[l]);

  // SA stages
  const float* pin = coord; const float* xin = feat;
  int nPrev = N_, cPrev = FEAT_;
  const int NS = 32;
  for (int s = 0; s < 4; ++s) {
    int np = NP[s], l0 = s * 3;
    k_fps<<<dim3(B_), dim3(256), 0, stream>>>(pin, fpsd, idxL[s], nPrev, np);
    k_gather_centers<<<elt(B_ * np), dim3(256), 0, stream>>>(pin, idxL[s], pL[s], nPrev, np);
    k_knn_group<<<dim3(B_ * np), dim3(256), 0, stream>>>(pin, xin, pL[s], actA,
                                                         nPrev, np, NS, cPrev, LKP[l0]);
    int M = B_ * np * NS;
    run_gemm(actA, l0,     M); bn_bf16(l0,     actB, M, LKP[l0 + 1], 1);
    run_gemm(actB, l0 + 1, M); bn_bf16(l0 + 1, actA, M, LKP[l0 + 2], 1);
    run_gemm(actA, l0 + 2, M); bn_f32(l0 + 2, outF, M);
    k_maxpool<<<elt(B_ * np * XC[s]), dim3(256), 0, stream>>>(outF, xL[s], B_ * np, NS, XC[s]);
    pin = pL[s]; xin = xL[s]; nPrev = np; cPrev = XC[s];
  }

  // FP4: (p3,x3) <- (p4,x4)
  { int M = B_ * 64;
    k_interp_concat<<<elt(B_ * 64), dim3(256), 0, stream>>>(pL[2], xL[2], pL[3], xL[3],
                                                            actA, 64, 16, 256, 512, LKP[12]);
    run_gemm(actA, 12, M); bn_bf16(12, actB, M, LKP[13], 1);
    run_gemm(actB, 13, M); bn_f32(13, x3n, M); }
  // FP3: (p2,x2) <- (p3,x3n)
  { int M = B_ * 256;
    k_interp_concat<<<elt(B_ * 256), dim3(256), 0, stream>>>(pL[1], xL[1], pL[2], x3n,
                                                             actA, 256, 64, 128, 256, LKP[14]);
    run_gemm(actA, 14, M); bn_bf16(14, actB, M, LKP[15], 1);
    run_gemm(actB, 15, M); bn_f32(15, x2n, M); }
  // FP2: (p1,x1) <- (p2,x2n)
  { int M = B_ * 1024;
    k_interp_concat<<<elt(B_ * 1024), dim3(256), 0, stream>>>(pL[0], xL[0], pL[1], x2n,
                                                              actA, 1024, 256, 64, 256, LKP[16]);
    run_gemm(actA, 16, M); bn_bf16(16, actB, M, LKP[17], 1);
    run_gemm(actB, 17, M); bn_f32(17, x1n, M); }
  // FP1: (p0, none) <- (p1,x1n), 3 layers of 128
  { int M = B_ * N_;
    k_interp_concat<<<elt(B_ * N_), dim3(256), 0, stream>>>(coord, nullptr, pL[0], x1n,
                                                            actA, N_, 1024, 0, 128, LKP[18]);
    run_gemm(actA, 18, M); bn_bf16(18, actB, M, 128, 1);
    run_gemm(actB, 19, M); bn_bf16(19, actA, M, 128, 1);
    run_gemm(actA, 20, M); bn_bf16(20, actB, M, 128, 1);   // x0n as bf16 head input
    // head: GEMM(128x128, bias cancels in BN) -> BN (no ReLU) -> 13-way linear
    run_gemm(actB, 21, M); bn_bf16(21, actA, M, 128, 0);
    k_head_out<<<elt(M * NCLS_), dim3(256), 0, stream>>>(actA, head_w2, head_b2,
                                                         (float*)d_out, M); }
}
